// RITS_61933428408512
// MI455X (gfx1250) — matrix operand order repaired and checked by reading, not compiled, not run on hardware
//
#include <hip/hip_runtime.h>
#include <stdint.h>

// ---------------------------------------------------------------------------
// RITS forward for MI455X (gfx1250), bf16 WMMA 16x16x32, f32 accumulate.
// B=512, T=96, F=128, H=512. Strategy:
//   pre-pass: convert/transpose weights to bf16, compute delta (elementwise
//             recursion), gamma_x/concat-A (elementwise), and gamma_h as ONE
//             big GEMM over M = B*T = 49152 rows (fully parallel).
//   scan:     96 x 4 fused kernels; weights (~3.2MB bf16) stay in 192MB L2.
// ---------------------------------------------------------------------------

#define B_  512
#define T_  96
#define F_  128
#define H_  512
#define BTF_ ((size_t)B_ * T_ * F_)

typedef __attribute__((ext_vector_type(16))) __bf16 v16bf;
typedef __attribute__((ext_vector_type(8)))  float  v8f;

union FragU { v16bf v; uint32_t d[8]; unsigned short u[16]; };

__device__ __forceinline__ unsigned short f2bf(float x) {
  union { float f; uint32_t u; } c; c.f = x;
  uint32_t u = c.u;
  uint32_t r = u + 0x7fffu + ((u >> 16) & 1u);          // round-to-nearest-even
  if ((u & 0x7f800000u) == 0x7f800000u) r = u;          // inf/nan passthrough
  return (unsigned short)(r >> 16);
}
__device__ __forceinline__ float bf2f(unsigned short h) {
  union { uint32_t u; float f; } c; c.u = ((uint32_t)h) << 16; return c.f;
}
__device__ __forceinline__ float sigm(float x) { return 1.f / (1.f + __expf(-x)); }

// LDS pitches (ushort elems). 34 keeps dword alignment, breaks bank-stride.
#define LDA 34
#define LDB 34

// Stage a 64x32 bf16 activation tile (row-major, rowStride elems) into LDS.
__device__ __forceinline__ void stageA64(unsigned short* sA, const unsigned short* g,
                                         size_t rowBase, size_t rowStride, int k0) {
  int tid = threadIdx.x;
#pragma unroll
  for (int i = 0; i < 4; i++) {
    int lin = tid + i * 256;          // 1024 dwords = 64 rows x 16 dwords
    int r = lin >> 4;
    int c = (lin & 15) * 2;
    uint32_t v = *(const uint32_t*)(g + (rowBase + r) * rowStride + k0 + c);
    *(uint32_t*)(sA + r * LDA + c) = v;
  }
}

// Stage a 32(K)x32(N) bf16 weight tile from W^T [K x N] row-major, TRANSPOSED
// into LDS as Bt[n][k] so B-fragment reads are K-contiguous per lane.
__device__ __forceinline__ void stageBt32(unsigned short* sB, const unsigned short* wt,
                                          int pitch, int k0, int n0) {
  int tid = threadIdx.x;
#pragma unroll
  for (int i = 0; i < 4; i++) {
    int lin = tid + i * 256;          // 1024 elems
    int k = lin >> 5;
    int n = lin & 31;
    sB[n * LDB + k] = wt[(size_t)(k0 + k) * pitch + (n0 + n)];
  }
}

// A-matrix 16x32 bf16 fragment (ISA 7.12.2): lane L<16 row M=L holds K 0-7
// (v0-3) and 16-23 (v4-7); lane L+16 holds K 8-15 and 24-31.
__device__ __forceinline__ v16bf loadA(const unsigned short* sA, int row0) {
  int lane = threadIdx.x & 31;
  const unsigned short* p = sA + (row0 + (lane & 15)) * LDA;
  int half = (lane >> 4) * 8;
  FragU f;
#pragma unroll
  for (int j = 0; j < 4; j++) {
    f.d[j]     = *(const uint32_t*)(p + half + 2 * j);
    f.d[4 + j] = *(const uint32_t*)(p + 16 + half + 2 * j);
  }
  return f.v;
}

// B-matrix 32x16 bf16 fragment: lane L<16 col N=L holds K 0-7 (v0-3) and
// 16-23 (v4-7); lane L+16 holds K 8-15 and 24-31 (matches A layout).
__device__ __forceinline__ v16bf loadB(const unsigned short* sB, int col0) {
  int lane = threadIdx.x & 31;
  const unsigned short* p = sB + (col0 + (lane & 15)) * LDB + (lane >> 4) * 8;
  FragU f;
#pragma unroll
  for (int j = 0; j < 4; j++) {
    f.d[j]     = *(const uint32_t*)(p + 2 * j);
    f.d[4 + j] = *(const uint32_t*)(p + 16 + 2 * j);
  }
  return f.v;
}

#define WMMA_BF16(a, b, c) \
  __builtin_amdgcn_wmma_f32_16x16x32_bf16(false, (a), false, (b), (short)0, (c), false, false)

// ---------------------------------------------------------------------------
// Pre-pass kernels
// ---------------------------------------------------------------------------

__global__ void k_zero_state(float* h_f, float* c_f, unsigned short* h_bf) {
  int i = blockIdx.x * 256 + threadIdx.x;   // < B*H
  h_f[i] = 0.f; c_f[i] = 0.f; h_bf[i] = 0;
}

// dst = bf16(W^T): dst[k*N+n] = W[n*K+k]; zero_diag for FeatureRegression.
__global__ void k_wt(const float* src, unsigned short* dst, int N, int K, int zero_diag) {
  int e = blockIdx.x * 256 + threadIdx.x;
  if (e >= N * K) return;
  int k = e / N, n = e % N;
  float v = src[n * K + k];
  if (zero_diag && n == k) v = 0.f;
  dst[e] = f2bf(v);
}

// delta recursion per (b,f): delta[0]=0; delta[t]=m[t-1]+(1-m[t-1])*(delta[t-1]+1)
__global__ void k_delta(const int* mask, float* delta, unsigned short* dbf) {
  int i = blockIdx.x * 256 + threadIdx.x;   // b*F + f, < B*F
  int b = i >> 7, f = i & 127;
  const int* mp = mask + (size_t)b * T_ * F_ + f;
  float* dp = delta + (size_t)b * T_ * F_ + f;
  unsigned short* bp = dbf + (size_t)b * T_ * F_ + f;
  float d = 0.f;
  for (int t = 0; t < T_; t++) {
    dp[(size_t)t * F_] = d;
    bp[(size_t)t * F_] = f2bf(d);
    float m = (float)mp[(size_t)t * F_];
    d = m + (1.f - m) * (d + 1.f);
  }
}

// concatA[b*T+t] = [ gamma_x (F), m (F) ] in bf16; gamma_x uses diag(Wdx).
__global__ void k_concat(const float* delta, const int* mask, const float* Wdx,
                         const float* bdx, unsigned short* cA) {
  size_t e = (size_t)blockIdx.x * 256 + threadIdx.x;  // < B*T*F
  if (e >= BTF_) return;
  int f = (int)(e & 127);
  size_t row = e >> 7;                                 // b*T + t
  float gx = __expf(-fmaxf(delta[e] * Wdx[f * F_ + f] + bdx[f], 0.f));
  cA[row * 256 + f] = f2bf(gx);
  cA[row * 256 + 128 + f] = f2bf((float)mask[e]);
}

// gammaH = exp(-relu(delta @ Wdh^T + bdh)) over ALL timesteps at once.
// M = B*T = 49152, N = H = 512, K = F = 128.  grid (768, 16), 256 thr.
__global__ void k_gamma_h(const unsigned short* dbf, const unsigned short* WdhT,
                          const float* bdh, unsigned short* gammaH) {
  __shared__ unsigned short sA[64 * LDA];
  __shared__ unsigned short sB[32 * LDB];
  int m0 = blockIdx.x * 64, n0 = blockIdx.y * 32;
  int wave = threadIdx.x >> 5, lane = threadIdx.x & 31;
  int mw = (wave >> 1) * 16, nw = (wave & 1) * 16;
  v8f acc = {};
  for (int k0 = 0; k0 < F_; k0 += 32) {
    __syncthreads();
    stageA64(sA, dbf, m0, F_, k0);
    stageBt32(sB, WdhT, H_, k0, n0);
    __syncthreads();
    v16bf a = loadA(sA, mw), b = loadB(sB, nw);
    acc = WMMA_BF16(a, b, acc);
  }
  int n = n0 + nw + (lane & 15);
  float bias = bdh[n];
  int half8 = (lane >> 4) * 8;
#pragma unroll
  for (int i = 0; i < 8; i++) {
    int m = m0 + mw + i + half8;
    float v = acc[i] + bias;
    gammaH[(size_t)m * H_ + n] = f2bf(__expf(-fmaxf(v, 0.f)));
  }
}

// ---------------------------------------------------------------------------
// Per-timestep kernels
// ---------------------------------------------------------------------------

// x_h = h @ Wh^T + bh; x_c = m*x + (1-m)*x_h.  grid (8,4).
__global__ void k_xh(const unsigned short* hbf, const unsigned short* WhT, const float* bh_,
                     const float* x, const int* mask, int t,
                     float* out_xh, float* xh_ws, unsigned short* xc_ws) {
  __shared__ unsigned short sA[64 * LDA];
  __shared__ unsigned short sB[32 * LDB];
  int m0 = blockIdx.x * 64, n0 = blockIdx.y * 32;
  int wave = threadIdx.x >> 5, lane = threadIdx.x & 31;
  int mw = (wave >> 1) * 16, nw = (wave & 1) * 16;
  v8f acc = {};
  for (int k0 = 0; k0 < H_; k0 += 32) {
    __syncthreads();
    stageA64(sA, hbf, m0, H_, k0);
    stageBt32(sB, WhT, F_, k0, n0);
    __syncthreads();
    v16bf a = loadA(sA, mw), b = loadB(sB, nw);
    acc = WMMA_BF16(a, b, acc);
  }
  int n = n0 + nw + (lane & 15);
  float bias = bh_[n];
  int half8 = (lane >> 4) * 8;
#pragma unroll
  for (int i = 0; i < 8; i++) {
    int m = m0 + mw + i + half8;
    float xh = acc[i] + bias;
    size_t io = ((size_t)m * T_ + t) * F_ + n;
    float mf = (float)mask[io];
    float xc = mf * x[io] + (1.f - mf) * xh;
    out_xh[io] = xh;
    xh_ws[m * F_ + n] = xh;
    xc_ws[m * F_ + n] = f2bf(xc);
  }
}

// h_dec = h * gammaH[t]  (bf16 for the gates GEMM).  grid 1024.
__global__ void k_hdec(const float* h_f, const unsigned short* gammaH, int t,
                       unsigned short* hdec) {
  int i = blockIdx.x * 256 + threadIdx.x;  // < B*H
  int b = i >> 9, n = i & 511;
  size_t gi = ((size_t)b * T_ + t) * H_ + n;
  hdec[i] = f2bf(h_f[i] * bf2f(gammaH[gi]));
}

// z_h = x_c@Wf^T+bf ; alpha = sigm(concatA@Wc^T+bc) ; c_h, c_c; build LSTM
// input matrix [c_c | m] in bf16.  grid (8,4).
__global__ void k_s2(const unsigned short* xc, const unsigned short* WfT, const float* bf_,
                     const unsigned short* cA, const unsigned short* WcT, const float* bc_,
                     const float* xh_ws, const float* x, const int* mask, int t,
                     float* out, unsigned short* inputsA) {
  __shared__ unsigned short sA[64 * LDA];
  __shared__ unsigned short sB[32 * LDB];
  int m0 = blockIdx.x * 64, n0 = blockIdx.y * 32;
  int wave = threadIdx.x >> 5, lane = threadIdx.x & 31;
  int mw = (wave >> 1) * 16, nw = (wave & 1) * 16;
  v8f accZ = {}, accA = {};
  for (int k0 = 0; k0 < F_; k0 += 32) {               // z_h, K=128
    __syncthreads();
    stageA64(sA, xc, m0, F_, k0);
    stageBt32(sB, WfT, F_, k0, n0);
    __syncthreads();
    v16bf a = loadA(sA, mw), b = loadB(sB, nw);
    accZ = WMMA_BF16(a, b, accZ);
  }
  const unsigned short* cAt = cA + (size_t)t * 256;   // row r = b*T + t
  for (int k0 = 0; k0 < 256; k0 += 32) {              // alpha, K=256
    __syncthreads();
    stageA64(sA, cAt, m0, (size_t)T_ * 256, k0);
    stageBt32(sB, WcT, F_, k0, n0);
    __syncthreads();
    v16bf a = loadA(sA, mw), b = loadB(sB, nw);
    accA = WMMA_BF16(a, b, accA);
  }
  int n = n0 + nw + (lane & 15);
  float bz = bf_[n], ba = bc_[n];
  int half8 = (lane >> 4) * 8;
#pragma unroll
  for (int i = 0; i < 8; i++) {
    int m = m0 + mw + i + half8;
    float z  = accZ[i] + bz;
    float al = sigm(accA[i] + ba);
    float xh = xh_ws[m * F_ + n];
    float ch = al * z + (1.f - al) * xh;
    size_t io = ((size_t)m * T_ + t) * F_ + n;
    float mf = (float)mask[io];
    float cc = mf * x[io] + (1.f - mf) * ch;
    out[io]             = cc;   // imputation
    out[BTF_ + io]      = ch;   // c_h
    out[2 * BTF_ + io]  = z;    // z_h
    inputsA[m * 256 + n]       = f2bf(cc);
    inputsA[m * 256 + 128 + n] = f2bf(mf);
  }
}

// gates = [c_c|m]@Wih^T + h_dec@Whh^T + bih + bhh, fused LSTM update.
// Block tile: 64 rows x 32 H-units, computed for ALL FOUR gate column groups
// (i,f,g,o) sharing one A fragment (4x A reuse).  grid (8,16).
__global__ void k_s3(const unsigned short* inA, const unsigned short* WihT,
                     const unsigned short* hdec, const unsigned short* WhhT,
                     const float* bih, const float* bhh,
                     float* c_f, float* h_f, unsigned short* h_bf) {
  __shared__ unsigned short sA[64 * LDA];
  __shared__ unsigned short sB[4][32 * LDB];
  int m0 = blockIdx.x * 64, n0 = blockIdx.y * 32;
  int wave = threadIdx.x >> 5, lane = threadIdx.x & 31;
  int mw = (wave >> 1) * 16, nw = (wave & 1) * 16;
  v8f acc[4] = {};
  for (int k0 = 0; k0 < 256; k0 += 32) {              // inputs phase, K=256
    __syncthreads();
    stageA64(sA, inA, m0, 256, k0);
#pragma unroll
    for (int g = 0; g < 4; g++) stageBt32(sB[g], WihT, 4 * H_, k0, g * H_ + n0);
    __syncthreads();
    v16bf a = loadA(sA, mw);
#pragma unroll
    for (int g = 0; g < 4; g++) {
      v16bf b = loadB(sB[g], nw);
      acc[g] = WMMA_BF16(a, b, acc[g]);
    }
  }
  for (int k0 = 0; k0 < H_; k0 += 32) {               // recurrent phase, K=512
    __syncthreads();
    stageA64(sA, hdec, m0, H_, k0);
#pragma unroll
    for (int g = 0; g < 4; g++) stageBt32(sB[g], WhhT, 4 * H_, k0, g * H_ + n0);
    __syncthreads();
    v16bf a = loadA(sA, mw);
#pragma unroll
    for (int g = 0; g < 4; g++) {
      v16bf b = loadB(sB[g], nw);
      acc[g] = WMMA_BF16(a, b, acc[g]);
    }
  }
  int n = n0 + nw + (lane & 15);
  float bI = bih[n]          + bhh[n];
  float bF = bih[H_ + n]     + bhh[H_ + n];
  float bG = bih[2 * H_ + n] + bhh[2 * H_ + n];
  float bO = bih[3 * H_ + n] + bhh[3 * H_ + n];
  int half8 = (lane >> 4) * 8;
#pragma unroll
  for (int i = 0; i < 8; i++) {
    int m = m0 + mw + i + half8;
    size_t idx = (size_t)m * H_ + n;
    float ig = sigm(acc[0][i] + bI);
    float fg = sigm(acc[1][i] + bF);
    float gg = tanhf(acc[2][i] + bG);
    float og = sigm(acc[3][i] + bO);
    float cn = fg * c_f[idx] + ig * gg;
    float hn = og * tanhf(cn);
    c_f[idx] = cn;
    h_f[idx] = hn;
    h_bf[idx] = f2bf(hn);
  }
}

// ---------------------------------------------------------------------------
// Host launcher
// ---------------------------------------------------------------------------

extern "C" void kernel_launch(void* const* d_in, const int* in_sizes, int n_in,
                              void* d_out, int out_size, void* d_ws, size_t ws_size,
                              hipStream_t stream) {
  (void)in_sizes; (void)n_in; (void)out_size; (void)ws_size;
  const float* x    = (const float*)d_in[0];
  const int*   mask = (const int*)d_in[1];
  const float* Wdh  = (const float*)d_in[2];
  const float* bdh  = (const float*)d_in[3];
  const float* Wdx  = (const float*)d_in[4];
  const float* bdx  = (const float*)d_in[5];
  const float* Wh   = (const float*)d_in[6];
  const float* bh   = (const float*)d_in[7];
  const float* Wf   = (const float*)d_in[8];
  const float* bf_  = (const float*)d_in[9];
  const float* Wc   = (const float*)d_in[10];
  const float* bc_  = (const float*)d_in[11];
  const float* Wih  = (const float*)d_in[12];
  const float* Whh  = (const float*)d_in[13];
  const float* bih  = (const float*)d_in[14];
  const float* bhh  = (const float*)d_in[15];
  float* out = (float*)d_out;

  char* base = (char*)d_ws;
  size_t off = 0;
  auto carve = [&](size_t bytes) -> char* {
    char* p = base + off;
    off += (bytes + 255) & ~(size_t)255;
    return p;
  };
  unsigned short* WdhT = (unsigned short*)carve((size_t)F_ * H_ * 2);        // [128][512]
  unsigned short* WhT  = (unsigned short*)carve((size_t)H_ * F_ * 2);        // [512][128]
  unsigned short* WfT  = (unsigned short*)carve((size_t)F_ * F_ * 2);        // [128][128]
  unsigned short* WcT  = (unsigned short*)carve((size_t)2 * F_ * F_ * 2);    // [256][128]
  unsigned short* WihT = (unsigned short*)carve((size_t)2 * F_ * 4 * H_ * 2);// [256][2048]
  unsigned short* WhhT = (unsigned short*)carve((size_t)H_ * 4 * H_ * 2);    // [512][2048]
  float*          dlt  = (float*)carve(BTF_ * 4);
  unsigned short* dbf  = (unsigned short*)carve(BTF_ * 2);
  unsigned short* gH   = (unsigned short*)carve((size_t)B_ * T_ * H_ * 2);
  unsigned short* cA   = (unsigned short*)carve((size_t)B_ * T_ * 2 * F_ * 2);
  float*          h_f  = (float*)carve((size_t)B_ * H_ * 4);
  float*          c_f  = (float*)carve((size_t)B_ * H_ * 4);
  unsigned short* h_bf = (unsigned short*)carve((size_t)B_ * H_ * 2);
  unsigned short* hdec = (unsigned short*)carve((size_t)B_ * H_ * 2);
  float*          xh_w = (float*)carve((size_t)B_ * F_ * 4);
  unsigned short* xc_w = (unsigned short*)carve((size_t)B_ * F_ * 2);
  unsigned short* inA  = (unsigned short*)carve((size_t)B_ * 2 * F_ * 2);

  // --- pre-pass ---
  k_zero_state<<<(B_ * H_) / 256, 256, 0, stream>>>(h_f, c_f, h_bf);
  k_wt<<<(H_ * F_ + 255) / 256, 256, 0, stream>>>(Wdh, WdhT, H_, F_, 0);
  k_wt<<<(F_ * H_ + 255) / 256, 256, 0, stream>>>(Wh,  WhT,  F_, H_, 0);
  k_wt<<<(F_ * F_ + 255) / 256, 256, 0, stream>>>(Wf,  WfT,  F_, F_, 1);
  k_wt<<<(F_ * 2 * F_ + 255) / 256, 256, 0, stream>>>(Wc, WcT, F_, 2 * F_, 0);
  k_wt<<<(4 * H_ * 2 * F_ + 255) / 256, 256, 0, stream>>>(Wih, WihT, 4 * H_, 2 * F_, 0);
  k_wt<<<(4 * H_ * H_ + 255) / 256, 256, 0, stream>>>(Whh, WhhT, 4 * H_, H_, 0);
  k_delta<<<(B_ * F_) / 256, 256, 0, stream>>>(mask, dlt, dbf);
  k_concat<<<(unsigned)((BTF_ + 255) / 256), 256, 0, stream>>>(dlt, mask, Wdx, bdx, cA);
  k_gamma_h<<<dim3((B_ * T_) / 64, H_ / 32), 256, 0, stream>>>(dbf, WdhT, bdh, gH);

  // --- sequential scan ---
  for (int t = 0; t < T_; t++) {
    k_xh<<<dim3(B_ / 64, F_ / 32), 256, 0, stream>>>(
        h_bf, WhT, bh, x, mask, t, out + 3 * BTF_, xh_w, xc_w);
    k_hdec<<<(B_ * H_) / 256, 256, 0, stream>>>(h_f, gH, t, hdec);
    k_s2<<<dim3(B_ / 64, F_ / 32), 256, 0, stream>>>(
        xc_w, WfT, bf_, cA, WcT, bc_, xh_w, x, mask, t, out, inA);
    k_s3<<<dim3(B_ / 64, H_ / 32), 256, 0, stream>>>(
        inA, WihT, hdec, WhhT, bih, bhh, c_f, h_f, h_bf);
  }
}
